// CausalSelfAttention_9431748182228
// MI455X (gfx1250) — matrix operand order repaired and checked by reading, not compiled, not run on hardware
//
#include <hip/hip_runtime.h>

// ---------------------------------------------------------------------------
// CDNA5 / gfx1250 causal self-attention:  y = proj(softmax(mask(QK^T))V)
// bf16 WMMA (v_wmma_f32_16x16x32_bf16) everywhere, f32 accumulate.
// ---------------------------------------------------------------------------

typedef __bf16 bf16;
typedef __attribute__((ext_vector_type(8)))  __bf16 v8bf;
typedef __attribute__((ext_vector_type(16))) __bf16 v16bf;
typedef __attribute__((ext_vector_type(8)))  float  v8f;

#define WMMA_BF16(A, B, C) \
    __builtin_amdgcn_wmma_f32_16x16x32_bf16(false, (A), false, (B), (short)0, (C), false, false)

static constexpr int BATCH = 2;
static constexpr int SEQ   = 2048;
static constexpr int DM    = 1024;
static constexpr int NH    = 16;
static constexpr int DH    = 64;
static constexpr int ROWS  = BATCH * SEQ;   // 4096

// A-operand fragment (16x32 bf16, row = lane%16, K contiguous 16h .. 16h+15)
__device__ inline v16bf load_fragA(const bf16* __restrict__ p) {
    v8bf lo = *(const v8bf*)p;         // K = 16h .. 16h+7
    v8bf hi = *(const v8bf*)(p + 8);   // K = 16h+8 .. 16h+15
    return __builtin_shufflevector(lo, hi, 0,1,2,3,4,5,6,7,8,9,10,11,12,13,14,15);
}

// ---------------------------------------------------------------------------
// elementwise prep kernels
// ---------------------------------------------------------------------------
__global__ __launch_bounds__(256) void cvt_f32_bf16_kernel(
    const float* __restrict__ in, bf16* __restrict__ out, int n) {
    int i = blockIdx.x * 256 + threadIdx.x;
    if (i < n) out[i] = (bf16)in[i];
}

// in: K x N (f32, row major)  ->  out: N x K (bf16, row major)  == B^T
__global__ __launch_bounds__(256) void transpose_to_bf16_kernel(
    const float* __restrict__ in, bf16* __restrict__ out, int K, int N) {
    int i = blockIdx.x * 256 + threadIdx.x;
    if (i >= K * N) return;
    int nI = i / K;
    int kI = i - nI * K;
    out[i] = (bf16)in[(size_t)kI * N + nI];
}

// qkvb [4096][3072] -> Q[b,h,s,d], K[b,h,s,d], Vt[b,h,d,s]
__global__ __launch_bounds__(256) void reshuffle_qkv_kernel(
    const bf16* __restrict__ qkvb, bf16* __restrict__ Qb,
    bf16* __restrict__ Kb, bf16* __restrict__ Vt) {
    int i = blockIdx.x * 256 + threadIdx.x;
    if (i >= ROWS * 3 * DM) return;
    int c   = i % (3 * DM);
    int tok = i / (3 * DM);
    int b   = tok >> 11;            // /2048
    int q   = tok & 2047;
    int which = c >> 10;            // 0:q 1:k 2:v
    int rr  = c & 1023;
    int h   = rr >> 6;
    int d   = rr & 63;
    int bh  = b * NH + h;
    bf16 v  = qkvb[i];
    if (which == 0)      Qb[((size_t)bh * SEQ + q) * DH + d] = v;
    else if (which == 1) Kb[((size_t)bh * SEQ + q) * DH + d] = v;
    else                 Vt[((size_t)bh * DH + d) * SEQ + q] = v;
}

// ---------------------------------------------------------------------------
// bf16 GEMM:  C[M x N] = A[M x K] * B[K x N],  B given as BT[N x K].
// Block: 256 threads = 8 waves (2x4), each wave -> 32x32 tile (2x2 WMMA).
// Block tile: 64 (M) x 128 (N). K stepped by 32.
// ---------------------------------------------------------------------------
template <bool OUT_BF16>
__global__ __launch_bounds__(256) void gemm_bf16_kernel(
    const bf16* __restrict__ A, const bf16* __restrict__ BT,
    void* __restrict__ Cout, int M, int N, int K) {
    const int lane = threadIdx.x & 31;
    const int wave = threadIdx.x >> 5;
    const int wm   = wave & 1;
    const int wn   = wave >> 1;
    const int r    = lane & 15;
    const int hh   = lane >> 4;

    const int mbase = blockIdx.x * 64  + wm * 32;
    const int nbase = blockIdx.y * 128 + wn * 32;

    v8f acc[2][2] = {};

    const bf16* a0 = A  + (size_t)(mbase + r)      * K + 16 * hh;
    const bf16* a1 = A  + (size_t)(mbase + 16 + r) * K + 16 * hh;
    const bf16* b0 = BT + (size_t)(nbase + r)      * K + 16 * hh;
    const bf16* b1 = BT + (size_t)(nbase + 16 + r) * K + 16 * hh;

    for (int k0 = 0; k0 < K; k0 += 32) {
        v16bf af0 = load_fragA(a0 + k0);
        v16bf af1 = load_fragA(a1 + k0);
        v16bf bf0 = *(const v16bf*)(b0 + k0);   // B frag: 16 contiguous K at 16h
        v16bf bf1 = *(const v16bf*)(b1 + k0);
        acc[0][0] = WMMA_BF16(af0, bf0, acc[0][0]);
        acc[0][1] = WMMA_BF16(af0, bf1, acc[0][1]);
        acc[1][0] = WMMA_BF16(af1, bf0, acc[1][0]);
        acc[1][1] = WMMA_BF16(af1, bf1, acc[1][1]);
    }

#pragma unroll
    for (int mt = 0; mt < 2; ++mt)
#pragma unroll
        for (int nt = 0; nt < 2; ++nt)
#pragma unroll
            for (int j = 0; j < 8; ++j) {
                int row = mbase + mt * 16 + j + 8 * hh;
                int col = nbase + nt * 16 + r;
                if (OUT_BF16)
                    ((bf16*)Cout)[(size_t)row * N + col] = (bf16)acc[mt][nt][j];
                else
                    ((float*)Cout)[(size_t)row * N + col] = acc[mt][nt][j];
            }
}

// ---------------------------------------------------------------------------
// Flash attention: one wave per (b, head, 16-query block).
// Per 32-key block: S = QK^T (4 WMMA), causal mask, online softmax
// (16-lane shfl_xor row reductions), P -> LDS transpose, O += P*V (4 WMMA).
// ---------------------------------------------------------------------------
__global__ __launch_bounds__(32) void attn_kernel(
    const bf16* __restrict__ Qb, const bf16* __restrict__ Kb,
    const bf16* __restrict__ Vt, bf16* __restrict__ Yb) {
    __shared__ bf16 Plds[16][40];          // 16x32 P tile, padded row stride

    const int lane  = threadIdx.x & 31;
    const int r     = lane & 15;
    const int hh    = lane >> 4;
    const int qblk  = blockIdx.x;          // 0..127
    const int head  = blockIdx.y;          // 0..15
    const int b     = blockIdx.z;          // 0..1
    const int bh    = b * NH + head;
    const int qbase = qblk * 16;

    // Q A-fragments, two d-chunks of 32
    const bf16* qrow = Qb + ((size_t)bh * SEQ + qbase + r) * DH;
    v16bf qf0 = load_fragA(qrow + 0  + 16 * hh);
    v16bf qf1 = load_fragA(qrow + 32 + 16 * hh);

    v8f o[4] = {};
    float mrow[8], lrow[8];
#pragma unroll
    for (int j = 0; j < 8; ++j) { mrow[j] = -1e30f; lrow[j] = 0.0f; }

    const float scale = 0.125f;            // 1/sqrt(64)

    for (int kbase = 0; kbase <= qbase + 15; kbase += 32) {
        // ---- S = Q K^T for 2 key sub-tiles of 16 ----
        v8f s0 = {}, s1 = {};
        {
            const bf16* krow = Kb + ((size_t)bh * SEQ + kbase + r) * DH;
            v16bf kf0 = *(const v16bf*)(krow + 0  + 16 * hh);
            v16bf kf1 = *(const v16bf*)(krow + 32 + 16 * hh);
            s0 = WMMA_BF16(qf0, kf0, s0);
            s0 = WMMA_BF16(qf1, kf1, s0);
        }
        {
            const bf16* krow = Kb + ((size_t)bh * SEQ + kbase + 16 + r) * DH;
            v16bf kf0 = *(const v16bf*)(krow + 0  + 16 * hh);
            v16bf kf1 = *(const v16bf*)(krow + 32 + 16 * hh);
            s1 = WMMA_BF16(qf0, kf0, s1);
            s1 = WMMA_BF16(qf1, kf1, s1);
        }

        // ---- causal mask + online softmax (rows j+8*hh, keys in lanes) ----
#pragma unroll
        for (int j = 0; j < 8; ++j) {
            const int qi = qbase + j + 8 * hh;
            float e0 = s0[j] * scale;
            float e1 = s1[j] * scale;
            if (kbase + r > qi)      e0 = -1e30f;
            if (kbase + 16 + r > qi) e1 = -1e30f;

            float t = fmaxf(e0, e1);
            t = fmaxf(t, __shfl_xor(t, 1, 32));
            t = fmaxf(t, __shfl_xor(t, 2, 32));
            t = fmaxf(t, __shfl_xor(t, 4, 32));
            t = fmaxf(t, __shfl_xor(t, 8, 32));

            const float mnew  = fmaxf(mrow[j], t);
            const float alpha = __expf(mrow[j] - mnew);
            mrow[j] = mnew;

            e0 = __expf(e0 - mnew);
            e1 = __expf(e1 - mnew);
            float rs = e0 + e1;
            rs += __shfl_xor(rs, 1, 32);
            rs += __shfl_xor(rs, 2, 32);
            rs += __shfl_xor(rs, 4, 32);
            rs += __shfl_xor(rs, 8, 32);
            lrow[j] = lrow[j] * alpha + rs;

#pragma unroll
            for (int dt = 0; dt < 4; ++dt) o[dt][j] *= alpha;

            Plds[j + 8 * hh][r]      = (bf16)e0;
            Plds[j + 8 * hh][16 + r] = (bf16)e1;
        }
        __syncthreads();   // single-wave WG: barrier -> S_NOP, orders LDS

        // ---- P as A-operand, O += P * V ----
        v8bf plo = *(const v8bf*)&Plds[r][16 * hh];
        v8bf phi = *(const v8bf*)&Plds[r][16 * hh + 8];
        v16bf pf = __builtin_shufflevector(plo, phi,
                       0,1,2,3,4,5,6,7,8,9,10,11,12,13,14,15);
#pragma unroll
        for (int dt = 0; dt < 4; ++dt) {
            const bf16* vrow =
                Vt + ((size_t)bh * DH + dt * 16 + r) * SEQ + kbase + 16 * hh;
            v16bf vf = *(const v16bf*)vrow;
            o[dt] = WMMA_BF16(pf, vf, o[dt]);
        }
        __syncthreads();
    }

    // ---- normalize + store y[b, q, head*64 + d] as bf16 ----
#pragma unroll
    for (int dt = 0; dt < 4; ++dt)
#pragma unroll
        for (int j = 0; j < 8; ++j) {
            float val = o[dt][j] / lrow[j];
            Yb[((size_t)b * SEQ + qbase + j + 8 * hh) * DM +
               head * DH + dt * 16 + r] = (bf16)val;
        }
}

// ---------------------------------------------------------------------------
// host-side launcher
// ---------------------------------------------------------------------------
extern "C" void kernel_launch(void* const* d_in, const int* in_sizes, int n_in,
                              void* d_out, int out_size, void* d_ws, size_t ws_size,
                              hipStream_t stream) {
    (void)in_sizes; (void)n_in; (void)out_size; (void)ws_size;
    const float* x      = (const float*)d_in[0];   // [2,2048,1024]
    const float* W_qkv  = (const float*)d_in[1];   // [1024,3072]
    const float* W_proj = (const float*)d_in[2];   // [1024,1024]
    float* out = (float*)d_out;                    // [2,2048,1024]

    char* base = (char*)d_ws;
    size_t off = 0;
    auto carve = [&](size_t bytes) {
        char* p = base + off;
        off += (bytes + 255) & ~(size_t)255;
        return p;
    };
    bf16* xb     = (bf16*)carve((size_t)ROWS * DM * 2);        // 8 MB
    bf16* wqkvT  = (bf16*)carve((size_t)3 * DM * DM * 2);      // 6 MB
    bf16* wprojT = (bf16*)carve((size_t)DM * DM * 2);          // 2 MB
    bf16* qkvb   = (bf16*)carve((size_t)ROWS * 3 * DM * 2);    // 24 MB
    bf16* Qb     = (bf16*)carve((size_t)ROWS * DM * 2);        // 8 MB
    bf16* Kb     = (bf16*)carve((size_t)ROWS * DM * 2);        // 8 MB
    bf16* Vt     = (bf16*)carve((size_t)ROWS * DM * 2);        // 8 MB
    bf16* Yb     = (bf16*)carve((size_t)ROWS * DM * 2);        // 8 MB

    // 1) bf16 conversions / weight transposes
    {
        int n = ROWS * DM;
        cvt_f32_bf16_kernel<<<n / 256, 256, 0, stream>>>(x, xb, n);
    }
    {
        int n = DM * 3 * DM;
        transpose_to_bf16_kernel<<<n / 256, 256, 0, stream>>>(W_qkv, wqkvT, DM, 3 * DM);
    }
    {
        int n = DM * DM;
        transpose_to_bf16_kernel<<<n / 256, 256, 0, stream>>>(W_proj, wprojT, DM, DM);
    }

    // 2) QKV GEMM: [4096 x 3072] = xb [4096 x 1024] * W_qkv
    {
        dim3 grid(ROWS / 64, (3 * DM) / 128);
        gemm_bf16_kernel<true><<<grid, 256, 0, stream>>>(xb, wqkvT, qkvb,
                                                         ROWS, 3 * DM, DM);
    }

    // 3) scatter into attention-friendly layouts
    {
        int n = ROWS * 3 * DM;
        reshuffle_qkv_kernel<<<n / 256, 256, 0, stream>>>(qkvb, Qb, Kb, Vt);
    }

    // 4) causal flash attention
    {
        dim3 grid(SEQ / 16, NH, BATCH);
        attn_kernel<<<grid, 32, 0, stream>>>(Qb, Kb, Vt, Yb);
    }

    // 5) output projection: out [4096 x 1024] = Yb * W_proj  (f32 result)
    {
        dim3 grid(ROWS / 64, DM / 128);
        gemm_bf16_kernel<false><<<grid, 256, 0, stream>>>(Yb, wprojT, out,
                                                          ROWS, DM, DM);
    }
}
